// involution2_52201032516022
// MI455X (gfx1250) — matrix operand order repaired and checked by reading, not compiled, not run on hardware
//
#include <hip/hip_runtime.h>

// ---------------------------------------------------------------------------
// Fused involution kernel for gfx1250 (MI455X).
//   prep kernel: fold BN, convert W1/W2 to f16 once into d_ws
//   main kernel: conv1 + conv2 via v_wmma_f32_16x16x32_f16 (f32 acc),
//                W2 staged to LDS via Tensor Data Mover (tensor_load_to_lds),
//                dynamic 7x7 depthwise aggregation from LDS with VALU FMAs.
// One 8x8 pixel tile per workgroup; ~215 KB of the 320 KB/WGP LDS used, so
// HBM traffic is just x + y + out (~50 MB total @ 23.3 TB/s).
// ---------------------------------------------------------------------------

typedef __attribute__((ext_vector_type(16))) _Float16 v16h;
typedef __attribute__((ext_vector_type(8)))  _Float16 v8h;
typedef __attribute__((ext_vector_type(8)))  float    v8f;
typedef __attribute__((ext_vector_type(4)))  unsigned v4u;
typedef __attribute__((ext_vector_type(8)))  int      v8i;
typedef __attribute__((ext_vector_type(4)))  int      v4i;

#define KS    7
#define PADK  3
#define CCH   64     // channels
#define GCC   4      // group channels
#define GRP   16     // groups
#define CRR   32     // reduced channels
#define KK    49     // KS*KS
#define MW    784    // KK*GRP (49*16 -> exactly 49 M-tiles of 16)
#define TP    64     // pixels per tile (8x8)
#define HALO  14     // 8 + KS - 1
#define HW    128

// LDS layout (bytes)
#define OFF_W2   0                        // 784*32  f16 = 50176
#define OFF_X    50176                    // 64*64   f16 =  8192  [px][ch]
#define OFF_T    58368                    // 64*32   f16 =  4096  [px][cr]
#define OFF_W1   62464                    // 32*64   f16 =  4096  [o][c]
#define OFF_Y    66560                    // 64*196  f32 = 50176  [ch][r*14+c]
#define OFF_WGT  116736                   // 64*784  f16 = 100352 [px][o]
#define OFF_SD   217088                   // 64 f32: s[0..31], d[32..63]
#define OFF_B2   217344                   // 784 f32 = 3136
#define SMEM_SZ  220480

// workspace layout (bytes)
#define WS_W2H   0                        // 25088 f16 = 50176
#define WS_W1H   50176                    // 2048  f16 =  4096
#define WS_SD    54272                    // 64 f32 = 256
#define WS_NEED  54528

// A fragment for v_wmma_f32_16x16x32_f16, A row-major [row][k]:
// lane L / L+16 hold row (L&15); elems e -> K = half*16+e:
// two adjacent 16-byte chunks.
__device__ inline v16h load_a_frag(const _Float16* A, int lda, int row0,
                                   int kbase, int lane) {
  const int half = (lane >> 4) & 1;
  const int r    = lane & 15;
  const _Float16* p = A + (row0 + r) * lda + kbase + half * 16;
  union { v16h v; v8h h[2]; } u;
  u.h[0] = *(const v8h*)(p);
  u.h[1] = *(const v8h*)(p + 8);
  return u.v;
}

// B fragment, B stored n-major [col][k]: lane holds col = lane&15,
// elems e -> K = half*16 + e: one contiguous 32-byte chunk.
__device__ inline v16h load_b_frag(const _Float16* Bt, int ldb, int col0,
                                   int kbase, int lane) {
  const int half = (lane >> 4) & 1;
  const int c    = lane & 15;
  return *(const v16h*)(Bt + (col0 + c) * ldb + kbase + half * 16);
}

// --------------------------- prep kernel -----------------------------------
__global__ __launch_bounds__(256) void involution_prep(
    const float* __restrict__ w1, const float* __restrict__ b1,
    const float* __restrict__ bn_g, const float* __restrict__ bn_b,
    const float* __restrict__ bn_m, const float* __restrict__ bn_v,
    const float* __restrict__ w2,
    _Float16* __restrict__ w2h, _Float16* __restrict__ w1h,
    float* __restrict__ sd)
{
  const int i = blockIdx.x * 256 + threadIdx.x;
  const int stride = gridDim.x * 256;
  for (int j = i; j < MW * CRR; j += stride) w2h[j] = (_Float16)w2[j];
  for (int j = i; j < CRR * CCH; j += stride) w1h[j] = (_Float16)w1[j];
  if (i < CRR) {
    float s = bn_g[i] * rsqrtf(bn_v[i] + 1e-5f);
    sd[i] = s;
    sd[CRR + i] = (b1[i] - bn_m[i]) * s + bn_b[i];
  }
}

// --------------------------- main kernel -----------------------------------
__global__ __launch_bounds__(256) void involution_fused(
    const float* __restrict__ x,  const float* __restrict__ y,
    const float* __restrict__ w1, const float* __restrict__ b1,
    const float* __restrict__ bn_g, const float* __restrict__ bn_b,
    const float* __restrict__ bn_m, const float* __restrict__ bn_v,
    const float* __restrict__ w2, const float* __restrict__ b2,
    const _Float16* __restrict__ w2h, const _Float16* __restrict__ w1h,
    const float* __restrict__ sd, int from_ws,
    float* __restrict__ out)
{
  extern __shared__ char smem[];
  _Float16* w2_lds = (_Float16*)(smem + OFF_W2);
  _Float16* x_lds  = (_Float16*)(smem + OFF_X);
  _Float16* t_lds  = (_Float16*)(smem + OFF_T);
  _Float16* w1_lds = (_Float16*)(smem + OFF_W1);
  float*    y_lds  = (float*)   (smem + OFF_Y);
  _Float16* wg_lds = (_Float16*)(smem + OFF_WGT);
  float*    sd_lds = (float*)   (smem + OFF_SD);
  float*    b2_lds = (float*)   (smem + OFF_B2);

  const int tid  = threadIdx.x;
  const int lane = tid & 31;
  const int wv   = tid >> 5;          // 8 waves

  const int bi   = blockIdx.x >> 8;   // 256 tiles per image
  const int trem = blockIdx.x & 255;
  const int h0   = (trem >> 4) << 3;
  const int w0   = (trem & 15) << 3;

  // ---- stage weights --------------------------------------------------
  if (from_ws) {
    if (wv == 0) {
      // Tensor Data Mover: DMA 50176 B of f16 W2 straight into LDS.
      // D#: count=1, type=2(image); 1-D tile, data_size=8B, 6272 elems.
      unsigned lds_w2 = (unsigned)(uintptr_t)(void*)w2_lds; // LDS offset
      unsigned long long ga = (unsigned long long)(uintptr_t)w2h;
      v4u g0 = { 1u,                               // count=1
                 lds_w2,                           // lds_addr
                 (unsigned)(ga & 0xffffffffu),
                 (unsigned)((ga >> 32) & 0x01ffffffu) | 0x80000000u }; // type=2
      v8i g1 = { 0x30000,                          // data_size=3 (8B)
                 (int)(6272u << 16),               // tensor_dim0.lo16
                 (int)0x00010000u,                 // tensor_dim0.hi16=0, tensor_dim1=1
                 (int)(6272u << 16),               // tensor_dim1.hi16=0, tile_dim0=6272
                 0,                                // tile_dim1=0, tile_dim2=0
                 6272,                             // tensor_dim0_stride.lo32
                 0, 0 };
      v4i z4 = { 0, 0, 0, 0 };
#if defined(__clang_major__) && __clang_major__ >= 23
      v8i z8 = { 0, 0, 0, 0, 0, 0, 0, 0 };
      __builtin_amdgcn_tensor_load_to_lds(g0, g1, z4, z4, z8, 0);
#else
      __builtin_amdgcn_tensor_load_to_lds(g0, g1, z4, z4, 0);
#endif
    }
    ((v8h*)w1_lds)[tid] = ((const v8h*)w1h)[tid];      // 2048 f16 in 256x16B
    if (tid < 2 * CRR) sd_lds[tid] = sd[tid];
  } else {
    // self-contained fallback (no workspace): convert in-kernel
    for (int i = tid; i < MW * CRR; i += 256) w2_lds[i] = (_Float16)w2[i];
    for (int i = tid; i < CRR * CCH; i += 256) w1_lds[i] = (_Float16)w1[i];
    if (tid < CRR) {
      float s = bn_g[tid] * rsqrtf(bn_v[tid] + 1e-5f);
      sd_lds[tid] = s;
      sd_lds[CRR + tid] = (b1[tid] - bn_m[tid]) * s + bn_b[tid];
    }
  }
  for (int i = tid; i < MW; i += 256) b2_lds[i] = b2[i];

  // ---- x tile -> LDS [px][ch] f16 (n-major B-operand layout) -----------
  for (int i = tid; i < TP * CCH; i += 256) {
    int ch = i >> 6;
    int px = i & 63;
    int hh = h0 + (px >> 3), ww = w0 + (px & 7);
    float v = x[(((size_t)bi * CCH + ch) * HW + hh) * HW + ww];
    x_lds[px * CCH + ch] = (_Float16)v;
  }

  // ---- y halo -> LDS f32 [ch][14*14], zero-padded ----------------------
  for (int i = tid; i < CCH * HALO * HALO; i += 256) {
    int ch  = i / (HALO * HALO);
    int rem = i - ch * (HALO * HALO);
    int r   = rem / HALO, cc = rem - r * HALO;
    int hh = h0 - PADK + r, ww = w0 - PADK + cc;
    float v = 0.0f;
    if ((unsigned)hh < (unsigned)HW && (unsigned)ww < (unsigned)HW)
      v = y[(((size_t)bi * CCH + ch) * HW + hh) * HW + ww];
    y_lds[i] = v;
  }
  if (from_ws && wv == 0) __builtin_amdgcn_s_wait_tensorcnt(0);
  __syncthreads();

  // ---- GEMM1: t(32x64) = relu(s*(W1 @ x) + d) --------------------------
  {
    const int mt = wv & 1;            // 2 M-tiles x 4 N-tiles = 8 waves
    const int nt = wv >> 1;
    v8f acc = {};
    #pragma unroll
    for (int kb = 0; kb < CCH; kb += 32) {
      v16h a = load_a_frag(w1_lds, CCH, mt * 16, kb, lane);
      v16h b = load_b_frag(x_lds,  CCH, nt * 16, kb, lane);
      acc = __builtin_amdgcn_wmma_f32_16x16x32_f16(false, a, false, b,
                                                   (short)0, acc, false, false);
    }
    const int half = (lane >> 4) & 1;
    const int px   = nt * 16 + (lane & 15);
    const int m0   = mt * 16 + half * 8;
    v8h hv;
    #pragma unroll
    for (int e = 0; e < 8; ++e) {
      int m = m0 + e;
      float v = acc[e] * sd_lds[m] + sd_lds[CRR + m];
      hv[e] = (_Float16)(v > 0.0f ? v : 0.0f);
    }
    *(v8h*)(t_lds + px * CRR + m0) = hv;    // [px][cr], B-layout for GEMM2
  }
  __syncthreads();

  // ---- GEMM2: weight(784x64) = W2 @ t + b2 -> wg_lds[px][o] f16 --------
  {
    const int half = (lane >> 4) & 1;
    const int col  = lane & 15;
    #pragma unroll
    for (int nt = 0; nt < 4; ++nt) {
      v16h bfr = load_b_frag(t_lds, CRR, nt * 16, 0, lane); // hoisted per nt
      const int px = nt * 16 + col;
      for (int mt = wv; mt < 49; mt += 8) {                 // no div/mod
        v16h a = load_a_frag(w2_lds, CRR, mt * 16, 0, lane);
        v8f acc = {};
        acc = __builtin_amdgcn_wmma_f32_16x16x32_f16(false, a, false, bfr,
                                                     (short)0, acc, false, false);
        const int m0 = mt * 16 + half * 8;
        v8h hv;
        #pragma unroll
        for (int e = 0; e < 8; ++e)
          hv[e] = (_Float16)(acc[e] + b2_lds[m0 + e]);
        *(v8h*)(wg_lds + px * MW + m0) = hv;
      }
    }
  }
  __syncthreads();

  // ---- dynamic depthwise aggregation -----------------------------------
  for (int task = tid; task < TP * GRP; task += 256) {
    const int px = task & 63;
    const int g  = task >> 6;
    const int pi = px >> 3, pj = px & 7;
    const _Float16* wp = wg_lds + px * MW + g * KK;
    const float* yb = y_lds + (g * GCC) * (HALO * HALO);
    float a0 = 0.f, a1 = 0.f, a2 = 0.f, a3 = 0.f;
    #pragma unroll
    for (int kr = 0; kr < KS; ++kr) {
      #pragma unroll
      for (int kc = 0; kc < KS; ++kc) {
        float wk = (float)wp[kr * KS + kc];
        int off = (pi + kr) * HALO + (pj + kc);
        a0 += wk * yb[0 * HALO * HALO + off];
        a1 += wk * yb[1 * HALO * HALO + off];
        a2 += wk * yb[2 * HALO * HALO + off];
        a3 += wk * yb[3 * HALO * HALO + off];
      }
    }
    const int hh = h0 + pi, ww = w0 + pj;
    size_t ob = (((size_t)bi * CCH + g * GCC) * HW + hh) * HW + ww;
    out[ob]               = a0;
    out[ob + 1 * HW * HW] = a1;
    out[ob + 2 * HW * HW] = a2;
    out[ob + 3 * HW * HW] = a3;
  }
}

extern "C" void kernel_launch(void* const* d_in, const int* in_sizes, int n_in,
                              void* d_out, int out_size, void* d_ws, size_t ws_size,
                              hipStream_t stream) {
  (void)in_sizes; (void)n_in; (void)out_size;
  const float* x    = (const float*)d_in[0];
  const float* y    = (const float*)d_in[1];
  const float* w1   = (const float*)d_in[2];
  const float* b1   = (const float*)d_in[3];
  const float* bn_g = (const float*)d_in[4];
  const float* bn_b = (const float*)d_in[5];
  const float* bn_m = (const float*)d_in[6];
  const float* bn_v = (const float*)d_in[7];
  const float* w2   = (const float*)d_in[8];
  const float* b2   = (const float*)d_in[9];
  float* out = (float*)d_out;

  hipFuncSetAttribute((const void*)involution_fused,
                      hipFuncAttributeMaxDynamicSharedMemorySize, SMEM_SZ);

  const int from_ws = (d_ws != nullptr && ws_size >= (size_t)WS_NEED) ? 1 : 0;
  _Float16* w2h = nullptr; _Float16* w1h = nullptr; float* sd = nullptr;
  if (from_ws) {
    char* ws = (char*)d_ws;
    w2h = (_Float16*)(ws + WS_W2H);
    w1h = (_Float16*)(ws + WS_W1H);
    sd  = (float*)   (ws + WS_SD);
    involution_prep<<<dim3(64), dim3(256), 0, stream>>>(
        w1, b1, bn_g, bn_b, bn_m, bn_v, w2, w2h, w1h, sd);
  }
  // 4 images * 16x16 tiles of 8x8 pixels
  involution_fused<<<dim3(4 * 16 * 16), dim3(256), SMEM_SZ, stream>>>(
      x, y, w1, b1, bn_g, bn_b, bn_m, bn_v, w2, b2,
      w2h, w1h, sd, from_ws, out);
}
